// GatedGraphConv_3272765080007
// MI455X (gfx1250) — matrix operand order repaired and checked by reading, not compiled, not run on hardware
//
#include <hip/hip_runtime.h>
#include <hip/hip_fp16.h>

typedef __attribute__((ext_vector_type(16))) _Float16 v16h;
typedef __attribute__((ext_vector_type(8)))  _Float16 v8h;
typedef __attribute__((ext_vector_type(4)))  _Float16 v4h;
typedef __attribute__((ext_vector_type(8)))  float    v8f;

#define NN 50000
#define NE 600000
#define DD 128
#define NT 4
#define NS 5

// ---------- WMMA fragment helpers (layouts per CDNA5 ISA 7.12.2) ----------

// A: row-major f16 [*, 128], tile rows m0..m0+15, K k0..k0+31.
// lane<16: M=lane,   K = k0+0..7  and k0+16..23
// lane>=16:M=lane-16,K = k0+8..15 and k0+24..31
__device__ __forceinline__ v16h load_a_frag(const _Float16* __restrict__ A,
                                            int m0, int k0, int lane) {
  int row = m0 + (lane & 15);
  int kb  = k0 + ((lane >> 4) << 3);
  const _Float16* p = A + (size_t)row * DD + kb;
  v8h lo = *(const v8h*)p;
  v8h hi = *(const v8h*)(p + 16);
  return __builtin_shufflevector(lo, hi, 0,1,2,3,4,5,6,7,8,9,10,11,12,13,14,15);
}

// B = W^T tile: W row-major f16 [F, 128]. Cols f0..f0+15, K k0..k0+31.
// lane<16:  N=lane,    K = k0+0..7  and k0+16..23 (row f0+N)
// lane>=16: N=lane-16, K = k0+8..15 and k0+24..31
__device__ __forceinline__ v16h load_b_frag(const _Float16* __restrict__ W,
                                            int f0, int k0, int lane) {
  int f  = f0 + (lane & 15);
  int ks = k0 + ((lane >> 4) << 3);
  const _Float16* p = W + (size_t)f * DD + ks;
  v8h lo = *(const v8h*)p;
  v8h hi = *(const v8h*)(p + 16);
  return __builtin_shufflevector(lo, hi, 0,1,2,3,4,5,6,7,8,9,10,11,12,13,14,15);
}

__device__ __forceinline__ v8f wmma_f16(v16h a, v16h b, v8f c) {
  return __builtin_amdgcn_wmma_f32_16x16x32_f16(false, a, false, b, (short)0, c,
                                                false, false);
}

// ---------- kernels ----------

__global__ void f2h_kernel(const float* __restrict__ in, _Float16* __restrict__ out,
                           int n) {
  int i = blockIdx.x * blockDim.x + threadIdx.x;
  int stride = gridDim.x * blockDim.x;
  for (; i < n; i += stride) out[i] = (_Float16)in[i];
}

__global__ void zero_kernel(float* __restrict__ p, int n) {
  int i = blockIdx.x * blockDim.x + threadIdx.x;
  int stride = gridDim.x * blockDim.x;
  for (; i < n; i += stride) p[i] = 0.0f;
}

// t[k] = h @ W_lin[k]^T + b_lin[k], stored f16. One wave per 16x16 output tile.
__global__ void transform_kernel(const _Float16* __restrict__ h16,
                                 const _Float16* __restrict__ Wlin16,
                                 const float* __restrict__ b_lin,
                                 _Float16* __restrict__ t16) {
  const int MT = NN / 16;   // 3125
  const int CT = DD / 16;   // 8
  int wave = (blockIdx.x * blockDim.x + threadIdx.x) >> 5;
  int lane = threadIdx.x & 31;
  if (wave >= NT * MT * CT) return;   // wave-uniform
  int k   = wave / (MT * CT);
  int rem = wave % (MT * CT);
  int m0  = (rem / CT) * 16;
  int f0  = (rem % CT) * 16;
  const _Float16* W = Wlin16 + (size_t)k * DD * DD;
  v8f c = {};
  #pragma unroll
  for (int kk = 0; kk < DD; kk += 32) {
    v16h a = load_a_frag(h16, m0, kk, lane);
    v16h b = load_b_frag(W, f0, kk, lane);
    c = wmma_f16(a, b, c);
  }
  int n  = lane & 15;
  int mb = (lane >> 4) * 8;
  float bias = b_lin[k * DD + f0 + n];
  _Float16* out = t16 + (size_t)k * NN * DD;
  #pragma unroll
  for (int r = 0; r < 8; ++r)
    out[(size_t)(m0 + mb + r) * DD + f0 + n] = (_Float16)(c[r] + bias);
}

// One wave per edge: gather t16[et,src], f32 atomic scatter-add into a32[dst].
__global__ void edge_kernel(const int* __restrict__ src, const int* __restrict__ dst,
                            const int* __restrict__ ety,
                            const _Float16* __restrict__ t16,
                            float* __restrict__ a32) {
  int e = (blockIdx.x * blockDim.x + threadIdx.x) >> 5;
  if (e >= NE) return;
  int lane = threadIdx.x & 31;
  int k = ety[e] - 1;
  const _Float16* trow = t16 + ((size_t)k * NN + src[e]) * DD;
  float* arow = a32 + (size_t)dst[e] * DD;
  int j = lane * 4;
  v4h v = *(const v4h*)(trow + j);
  atomicAdd(arow + j + 0, (float)v[0]);
  atomicAdd(arow + j + 1, (float)v[1]);
  atomicAdd(arow + j + 2, (float)v[2]);
  atomicAdd(arow + j + 3, (float)v[3]);
}

__device__ __forceinline__ float sigmoidf_(float x) {
  return 1.0f / (1.0f + __expf(-x));
}

// Fused GRU cell: 6 gate GEMM tiles via WMMA + elementwise, per 16x16 tile.
__global__ void gru_kernel(const _Float16* __restrict__ a16,
                           const _Float16* __restrict__ h16,
                           const float* __restrict__ h32,
                           const _Float16* __restrict__ wih16,
                           const _Float16* __restrict__ whh16,
                           const float* __restrict__ b_ih,
                           const float* __restrict__ b_hh,
                           float* __restrict__ hout32,
                           _Float16* __restrict__ hout16) {
  const int CT = DD / 16;  // 8
  int wave = (blockIdx.x * blockDim.x + threadIdx.x) >> 5;
  int lane = threadIdx.x & 31;
  if (wave >= (NN / 16) * CT) return;   // wave-uniform
  int m0 = (wave / CT) * 16;
  int c0 = (wave % CT) * 16;
  v8f ir = {}, iz = {}, in_ = {}, hr = {}, hz = {}, hn = {};
  #pragma unroll
  for (int kk = 0; kk < DD; kk += 32) {
    v16h aa = load_a_frag(a16, m0, kk, lane);
    v16h ah = load_a_frag(h16, m0, kk, lane);
    v16h b0 = load_b_frag(wih16, c0, kk, lane);
    v16h b1 = load_b_frag(wih16, DD + c0, kk, lane);
    v16h b2 = load_b_frag(wih16, 2 * DD + c0, kk, lane);
    ir  = wmma_f16(aa, b0, ir);
    iz  = wmma_f16(aa, b1, iz);
    in_ = wmma_f16(aa, b2, in_);
    v16h c0f = load_b_frag(whh16, c0, kk, lane);
    v16h c1f = load_b_frag(whh16, DD + c0, kk, lane);
    v16h c2f = load_b_frag(whh16, 2 * DD + c0, kk, lane);
    hr = wmma_f16(ah, c0f, hr);
    hz = wmma_f16(ah, c1f, hz);
    hn = wmma_f16(ah, c2f, hn);
  }
  int n   = lane & 15;
  int mb  = (lane >> 4) * 8;
  int col = c0 + n;
  float bir = b_ih[col], biz = b_ih[DD + col], bin = b_ih[2 * DD + col];
  float bhr = b_hh[col], bhz = b_hh[DD + col], bhn = b_hh[2 * DD + col];
  #pragma unroll
  for (int r = 0; r < 8; ++r) {
    int row = m0 + mb + r;
    float rv = sigmoidf_((ir[r] + bir) + (hr[r] + bhr));
    float zv = sigmoidf_((iz[r] + biz) + (hz[r] + bhz));
    float nv = tanhf((in_[r] + bin) + rv * (hn[r] + bhn));
    float hold = h32[(size_t)row * DD + col];
    float hnew = (1.0f - zv) * nv + zv * hold;
    hout32[(size_t)row * DD + col] = hnew;
    hout16[(size_t)row * DD + col] = (_Float16)hnew;
  }
}

// ---------- launch ----------

extern "C" void kernel_launch(void* const* d_in, const int* in_sizes, int n_in,
                              void* d_out, int out_size, void* d_ws, size_t ws_size,
                              hipStream_t stream) {
  const float* h0    = (const float*)d_in[0];
  const int*   src   = (const int*)d_in[1];
  const int*   dst   = (const int*)d_in[2];
  const int*   ety   = (const int*)d_in[3];
  const float* W_lin = (const float*)d_in[4];
  const float* b_lin = (const float*)d_in[5];
  const float* w_ih  = (const float*)d_in[6];
  const float* w_hh  = (const float*)d_in[7];
  const float* b_ih  = (const float*)d_in[8];
  const float* b_hh  = (const float*)d_in[9];

  const size_t ND = (size_t)NN * DD;
  char* ws = (char*)d_ws;
  auto carve = [&](size_t bytes) {
    char* p = ws;
    ws += (bytes + 255) & ~(size_t)255;
    return p;
  };
  _Float16* Wlin16 = (_Float16*)carve((size_t)NT * DD * DD * 2);
  _Float16* wih16  = (_Float16*)carve((size_t)3 * DD * DD * 2);
  _Float16* whh16  = (_Float16*)carve((size_t)3 * DD * DD * 2);
  _Float16* h16A   = (_Float16*)carve(ND * 2);
  _Float16* h16B   = (_Float16*)carve(ND * 2);
  _Float16* a16    = (_Float16*)carve(ND * 2);
  _Float16* t16    = (_Float16*)carve((size_t)NT * ND * 2);
  float*    hA     = (float*)carve(ND * 4);
  float*    hB     = (float*)carve(ND * 4);
  float*    a32    = (float*)carve(ND * 4);

  const int TB = 256;
  // one-time (per launch) f16 conversions of weights and initial h
  f2h_kernel<<<(NT * DD * DD + TB - 1) / TB, TB, 0, stream>>>(W_lin, Wlin16, NT * DD * DD);
  f2h_kernel<<<(3 * DD * DD + TB - 1) / TB, TB, 0, stream>>>(w_ih, wih16, 3 * DD * DD);
  f2h_kernel<<<(3 * DD * DD + TB - 1) / TB, TB, 0, stream>>>(w_hh, whh16, 3 * DD * DD);
  f2h_kernel<<<2048, TB, 0, stream>>>(h0, h16A, (int)ND);

  const float* hcur32 = h0;
  _Float16*    hcur16 = h16A;
  float* ping32[2] = { hA, hB };

  const int transform_waves = NT * (NN / 16) * (DD / 16);   // 100000
  const int gru_waves       = (NN / 16) * (DD / 16);        // 25000
  const int edge_blocks     = NE / (TB / 32);               // 75000

  for (int step = 0; step < NS; ++step) {
    transform_kernel<<<transform_waves / (TB / 32), TB, 0, stream>>>(
        hcur16, Wlin16, b_lin, t16);
    zero_kernel<<<2048, TB, 0, stream>>>(a32, (int)ND);
    edge_kernel<<<edge_blocks, TB, 0, stream>>>(src, dst, ety, t16, a32);
    f2h_kernel<<<2048, TB, 0, stream>>>(a32, a16, (int)ND);

    float*    hnext32 = (step == NS - 1) ? (float*)d_out : ping32[step & 1];
    _Float16* hnext16 = (hcur16 == h16A) ? h16B : h16A;
    gru_kernel<<<gru_waves / (TB / 32), TB, 0, stream>>>(
        a16, hcur16, hcur32, wih16, whh16, b_ih, b_hh, hnext32, hnext16);
    hcur32 = hnext32;
    hcur16 = hnext16;
  }
}
